// GraphAttentionLayer_rel_68856915689917
// MI455X (gfx1250) — matrix operand order repaired and checked by reading, not compiled, not run on hardware
//
#include <hip/hip_runtime.h>
#include <hip/hip_bf16.h>

#define NN      8192      // nodes
#define FEATS   256       // features
#define RR      65536     // relation instances
#define INR     64
#define MM      (1 << 20) // pair triples
#define KT      64        // K-tile width per barrier iteration

typedef __attribute__((ext_vector_type(16))) __bf16         v16bf;
typedef __attribute__((ext_vector_type(8)))  float          v8f;
typedef __attribute__((ext_vector_type(8)))  unsigned short u16x8;

union ABu { v16bf v; u16x8 h[2]; };

// ---------- helpers ----------
__device__ __forceinline__ unsigned short f2bf(float x) {
    union { float f; unsigned u; } c; c.f = x;
    unsigned r = c.u + 0x7FFFu + ((c.u >> 16) & 1u);   // round-to-nearest-even
    return (unsigned short)(r >> 16);
}
__device__ __forceinline__ unsigned pack2bf(float lo, float hi) {
    return (unsigned)f2bf(lo) | ((unsigned)f2bf(hi) << 16);
}
// order-preserving float->uint key (monotone for ALL floats), so atomicMax_u32 == float max
__device__ __forceinline__ unsigned keyenc(float x) {
    unsigned u = __float_as_uint(x);
    return (u & 0x80000000u) ? ~u : (u + 0x80000000u);
}
__device__ __forceinline__ float keydec(unsigned k) {
    unsigned u = (k & 0x80000000u) ? (k - 0x80000000u) : ~k;
    return __uint_as_float(u);
}

// ---------- 1) seq[r] = dot(rel[r,:], W[0,:]) ----------
__global__ __launch_bounds__(256) void rel_mv(const float* __restrict__ rel,
                                              const float* __restrict__ W,
                                              float* __restrict__ seq) {
    __shared__ float w[INR];
    if (threadIdx.x < INR) w[threadIdx.x] = W[threadIdx.x];   // W row 0
    __syncthreads();
    int r = blockIdx.x * 256 + threadIdx.x;
    const float4* rr = (const float4*)(rel + (size_t)r * INR);
    float s = 0.f;
#pragma unroll
    for (int k = 0; k < INR / 4; ++k) {
        float4 v = rr[k];
        s += v.x * w[4*k] + v.y * w[4*k+1] + v.z * w[4*k+2] + v.w * w[4*k+3];
    }
    seq[r] = s;
}

// ---------- 2) bf16 transposed copy of input: xT[f][j] ----------
__global__ __launch_bounds__(256) void transpose_bf16(const float* __restrict__ x,
                                                      unsigned short* __restrict__ xT) {
    size_t t = (size_t)blockIdx.x * 256 + threadIdx.x;   // t < FEATS*NN
    int f = (int)(t >> 13);          // / NN
    int j = (int)(t & (NN - 1));
    xT[t] = f2bf(x[(size_t)j * FEATS + f]);
}

// ---------- 3) comb = keyenc(adj)  (relu(logits)+adj starts as adj) ----------
__global__ __launch_bounds__(256) void enc_copy(const float4* __restrict__ adj4,
                                                uint4* __restrict__ comb4) {
    size_t t = (size_t)blockIdx.x * 256 + threadIdx.x;   // t < NN*NN/4
    float4 a = adj4[t];
    uint4 k;
    k.x = keyenc(a.x); k.y = keyenc(a.y); k.z = keyenc(a.z); k.w = keyenc(a.w);
    comb4[t] = k;
}

// ---------- 4) scatter-max of positive pair values ----------
__global__ __launch_bounds__(256) void scatter_max(const float* __restrict__ seq,
                                                   const int* __restrict__ ps,
                                                   const int* __restrict__ pd,
                                                   const int* __restrict__ pr,
                                                   unsigned* __restrict__ comb) {
    int t = blockIdx.x * 256 + threadIdx.x;              // MM is multiple of 256
    float v = seq[pr[t]];
    if (v > 0.f) {   // cells start at max(adj) with adj==0 at every pair position
        atomicMax(comb + (size_t)ps[t] * NN + pd[t], keyenc(v));
    }
}

// ---------- 5) per-row online softmax stats: mrow[i], sinv[i]=1/sum ----------
__global__ __launch_bounds__(256) void row_stats(const unsigned* __restrict__ comb,
                                                 float* __restrict__ mrow,
                                                 float* __restrict__ sinv) {
    int i = blockIdx.x;
    const uint4* cr = (const uint4*)(comb + (size_t)i * NN);
    float m = -INFINITY, s = 0.f;
    for (int q = threadIdx.x; q < NN / 4; q += 256) {    // 128-bit coalesced loads
        uint4 kk = cr[q];
        float v0 = keydec(kk.x), v1 = keydec(kk.y);
        float v2 = keydec(kk.z), v3 = keydec(kk.w);
        float vm = fmaxf(fmaxf(v0, v1), fmaxf(v2, v3));
        float nm = fmaxf(m, vm);
        s = s * __expf(m - nm) + __expf(v0 - nm) + __expf(v1 - nm)
                               + __expf(v2 - nm) + __expf(v3 - nm);
        m = nm;
    }
    __shared__ float sm[256], ss[256];
    sm[threadIdx.x] = m; ss[threadIdx.x] = s;
    __syncthreads();
    for (int w = 128; w > 0; w >>= 1) {
        if (threadIdx.x < (unsigned)w) {
            float m1 = sm[threadIdx.x], m2 = sm[threadIdx.x + w];
            float nm = fmaxf(m1, m2);
            ss[threadIdx.x] = ss[threadIdx.x] * __expf(m1 - nm)
                            + ss[threadIdx.x + w] * __expf(m2 - nm);
            sm[threadIdx.x] = nm;
        }
        __syncthreads();
    }
    if (threadIdx.x == 0) { mrow[i] = sm[0]; sinv[i] = 1.0f / ss[0]; }  // diag => s>0
}

// ---------- 6) fused softmax * input via v_wmma_f32_16x16x32_bf16, then ELU ----------
__global__ __launch_bounds__(256) void attn_matmul(const unsigned* __restrict__ comb,
                                                   const float* __restrict__ mrow,
                                                   const float* __restrict__ sinv,
                                                   const unsigned short* __restrict__ xT,
                                                   const float* __restrict__ bias,
                                                   float* __restrict__ out) {
    __shared__ __align__(16) unsigned short Atile[2][16 * KT];  // bf16 P-tile, double buffered

    const int i0    = blockIdx.x * 16;
    const int t     = threadIdx.x;
    const int wave  = t >> 5;
    const int lane  = t & 31;
    const int nlo   = lane & 15;
    const int khalf = lane >> 4;
    const int fbase = wave * 32;     // 8 waves x 32 features = 256

    // A-tile production: thread t owns row (t>>4), 4 consecutive cols (t&15)*4
    const int rowA = t >> 4;                 // 0..15
    const int c4   = (t & 15) * 4;           // 0..60
    const int giA  = i0 + rowA;
    const float mA = mrow[giA], sA = sinv[giA];
    const uint4* crA = (const uint4*)(comb + (size_t)giA * NN + c4);   // +j0/4 per iter

    // B operand base pointers (xT is [FEATS][NN] bf16, contiguous in K=j)
    const unsigned short* bpt0 = xT + (size_t)(fbase + nlo)      * NN + khalf * 8;
    const unsigned short* bpt1 = xT + (size_t)(fbase + 16 + nlo) * NN + khalf * 8;

    v8f acc0 = {}, acc1 = {};

    for (int kt = 0; kt < NN / KT; ++kt) {
        const int j0  = kt * KT;
        const int buf = kt & 1;

        // build normalized-softmax A tile in LDS: one b128 load + one packed b64 store
        uint4 kk = crA[j0 / 4];
        float p0 = __expf(keydec(kk.x) - mA) * sA;
        float p1 = __expf(keydec(kk.y) - mA) * sA;
        float p2 = __expf(keydec(kk.z) - mA) * sA;
        float p3 = __expf(keydec(kk.w) - mA) * sA;
        *(uint2*)&Atile[buf][rowA * KT + c4] = uint2{pack2bf(p0, p1), pack2bf(p2, p3)};
        __syncthreads();

#pragma unroll
        for (int ksub = 0; ksub < 2; ++ksub) {
            const int kc = ksub * 32;
            // A operand per ISA 16-bit A 16x32 layout:
            // lanes<16: K 0-7 in v0-3, 16-23 in v4-7 ; lanes>=16: K 8-15, 24-31
            ABu a;
            a.h[0] = *(const u16x8*)&Atile[buf][nlo * KT + kc + khalf * 8];
            a.h[1] = *(const u16x8*)&Atile[buf][nlo * KT + kc + 16 + khalf * 8];

            // B operand (32x16 K-major per column): lanes<16 hold K 0-7, 16-23 ; lanes>=16 K 8-15, 24-31
            ABu b0, b1;
            const unsigned short* p0p = bpt0 + j0 + kc;
            const unsigned short* p1p = bpt1 + j0 + kc;
            b0.h[0] = *(const u16x8*)(p0p);  b0.h[1] = *(const u16x8*)(p0p + 16);
            b1.h[0] = *(const u16x8*)(p1p);  b1.h[1] = *(const u16x8*)(p1p + 16);

            acc0 = __builtin_amdgcn_wmma_f32_16x16x32_bf16(false, a.v, false, b0.v,
                                                           (short)0, acc0, false, false);
            acc1 = __builtin_amdgcn_wmma_f32_16x16x32_bf16(false, a.v, false, b1.v,
                                                           (short)0, acc1, false, false);
        }
        // single barrier per iter is safe with double-buffered Atile
    }

    // C layout: lanes<16: M=v, N=lane ; lanes>=16: M=v+8, N=lane-16
    const int f0 = fbase + nlo;
    const float bb0 = bias[f0], bb1 = bias[f0 + 16];
#pragma unroll
    for (int v = 0; v < 8; ++v) {
        int row = i0 + v + khalf * 8;
        float x0 = acc0[v] + bb0;
        float x1 = acc1[v] + bb1;
        out[(size_t)row * FEATS + f0]      = (x0 > 0.f) ? x0 : expm1f(x0);
        out[(size_t)row * FEATS + f0 + 16] = (x1 > 0.f) ? x1 : expm1f(x1);
    }
}

// ---------- launch ----------
extern "C" void kernel_launch(void* const* d_in, const int* in_sizes, int n_in,
                              void* d_out, int out_size, void* d_ws, size_t ws_size,
                              hipStream_t stream) {
    const float* x    = (const float*)d_in[0];   // [N, FEAT]
    const float* rel  = (const float*)d_in[1];   // [R, IN_RELS]
    const float* W    = (const float*)d_in[2];   // [OUT_RELS, IN_RELS]
    const float* bias = (const float*)d_in[3];   // [FEAT]
    const float* adj  = (const float*)d_in[4];   // [N, N]
    const int*   ps   = (const int*)d_in[5];
    const int*   pd   = (const int*)d_in[6];
    const int*   pr   = (const int*)d_in[7];
    float* out = (float*)d_out;

    // workspace carve (all 256B-aligned chunks): ~260.3 MB total
    char* w = (char*)d_ws;
    float*          seq  = (float*)w;           w += (size_t)RR * 4;          // 256 KB
    float*          mrow = (float*)w;           w += (size_t)NN * 4;          // 32 KB
    float*          sinv = (float*)w;           w += (size_t)NN * 4;          // 32 KB
    unsigned short* xT   = (unsigned short*)w;  w += (size_t)FEATS * NN * 2;  // 4 MB
    unsigned*       comb = (unsigned*)w;        /* NN*NN*4 = 256 MB */

    rel_mv        <<<RR / 256, 256, 0, stream>>>(rel, W, seq);
    transpose_bf16<<<(FEATS * NN) / 256, 256, 0, stream>>>(x, xT);
    enc_copy      <<<((size_t)NN * NN / 4) / 256, 256, 0, stream>>>((const float4*)adj,
                                                                    (uint4*)comb);
    scatter_max   <<<MM / 256, 256, 0, stream>>>(seq, ps, pd, pr, comb);
    row_stats     <<<NN, 256, 0, stream>>>(comb, mrow, sinv);
    attn_matmul   <<<NN / 16, 256, 0, stream>>>(comb, mrow, sinv, xT, bias, out);
}
